// CausalSelfAttention_35802847379938
// MI455X (gfx1250) — compile-verified, matrix operand order checked by reading, not run on hardware
//
#include <hip/hip_runtime.h>
#include <hip/hip_bf16.h>
#include <cstdint>

typedef __attribute__((ext_vector_type(16))) _Float16 v16h;
typedef __attribute__((ext_vector_type(8)))  _Float16 v8h;
typedef __attribute__((ext_vector_type(8)))  float    v8f;

#define BATCH   4
#define SEQ     2048
#define CDIM    1024
#define NHEAD   16
#define HDIM    64

static __device__ __forceinline__ v16h frag_from(const _Float16* lo_p) {
    v8h lo = *(const v8h*)lo_p;
    v8h hi = *(const v8h*)(lo_p + 16);
    v16h f;
    #pragma unroll
    for (int e = 0; e < 8; ++e) { f[e] = lo[e]; f[e + 8] = hi[e]; }
    return f;
}

// ---------------------------------------------------------------------------
// Kernel 1: qkv = x @ W_attn + b_attn   (M=8192, N=3072, K=1024)
// Block tile 64x128, 8 waves: wave = 16m x 64n (4 wmma accumulators),
// K-step 32, double-buffered LDS (1 barrier/iter, staging overlaps wmma).
// Epilogue scatters into q[B,H,T,D] (x 1/sqrt(D)), k[B,H,T,D], vT[B,H,D,T].
// ---------------------------------------------------------------------------
__global__ __launch_bounds__(256) void qkv_gemm_kernel(
    const float* __restrict__ X,      // [8192,1024]
    const float* __restrict__ W,      // [1024,3072]
    const float* __restrict__ bias,   // [3072]
    _Float16* __restrict__ q16,
    _Float16* __restrict__ k16,
    _Float16* __restrict__ vT16)
{
    __shared__ _Float16 As[2][64 * 32];    // [m][k]
    __shared__ _Float16 Bs[2][128 * 32];   // [n][k]

    const int tid  = threadIdx.x;
    const int lane = tid & 31;
    const int wave = tid >> 5;
    const int wm   = wave & 3;           // 4 waves along M
    const int wn   = wave >> 2;          // 2 waves along N
    const int m0   = blockIdx.x * 64;
    const int n0   = blockIdx.y * 128;
    const int tl   = lane & 15;
    const int hh   = lane >> 4;
    const int K    = 1024, ldb = 3072;

    v8f acc[4];
    #pragma unroll
    for (int nt = 0; nt < 4; ++nt)
        #pragma unroll
        for (int r = 0; r < 8; ++r) acc[nt][r] = 0.0f;

    // per-thread staging coordinates
    const int am  = tid >> 2, akc = (tid & 3) * 8;       // A: 64x32, 8 elems
    const int bn  = tid & 127, bkc0 = (tid >> 7) * 16;   // B: 32x128, 16 elems
    const float* aSrc = X + (size_t)(m0 + am) * K + akc;
    const float* bSrc = W + n0 + bn;

    auto stage = [&](int k0, int p) {
        {   // A tile: two b128 global loads -> one b128 LDS store
            const float4* s = (const float4*)(aSrc + k0);
            float4 f0 = s[0], f1 = s[1];
            v8h pk;
            pk[0] = (_Float16)f0.x; pk[1] = (_Float16)f0.y;
            pk[2] = (_Float16)f0.z; pk[3] = (_Float16)f0.w;
            pk[4] = (_Float16)f1.x; pk[5] = (_Float16)f1.y;
            pk[6] = (_Float16)f1.z; pk[7] = (_Float16)f1.w;
            *(v8h*)&As[p][am * 32 + akc] = pk;
        }
        #pragma unroll
        for (int c = 0; c < 2; ++c) {   // B tile: lane-coalesced loads -> b128 LDS stores
            int kc = bkc0 + c * 8;
            const float* s = bSrc + (size_t)(k0 + kc) * ldb;
            v8h pk;
            #pragma unroll
            for (int i = 0; i < 8; ++i) pk[i] = (_Float16)s[(size_t)i * ldb];
            *(v8h*)&Bs[p][bn * 32 + kc] = pk;
        }
    };

    stage(0, 0);
    for (int k0 = 0; k0 < K; k0 += 32) {
        const int p = (k0 >> 5) & 1;
        __syncthreads();                         // buf p ready; buf p^1 free
        if (k0 + 32 < K) stage(k0 + 32, p ^ 1);  // overlaps with wmma below

        // preload all fragments first (distinct regs -> staged dscnt waits)
        v16h a = frag_from(&As[p][(wm * 16 + tl) * 32 + 8 * hh]);
        v16h bfr[4];
        #pragma unroll
        for (int nt = 0; nt < 4; ++nt)
            bfr[nt] = frag_from(&Bs[p][(wn * 64 + nt * 16 + tl) * 32 + 8 * hh]);
        #pragma unroll
        for (int nt = 0; nt < 4; ++nt)
            acc[nt] = __builtin_amdgcn_wmma_f32_16x16x32_f16(false, a, false, bfr[nt], (short)0, acc[nt], false, false);
    }

    // epilogue: scalarize q/k/v routing (wave-uniform) via readfirstlane
    const int gm0 = m0 + wm * 16;
    const int b   = __builtin_amdgcn_readfirstlane(gm0 >> 11);
    const int tb  = (gm0 & 2047) + 8 * hh;          // rows r: t = tb + r
    #pragma unroll
    for (int nt = 0; nt < 4; ++nt) {
        const int gnb   = n0 + wn * 64 + nt * 16;
        const int which = __builtin_amdgcn_readfirstlane(gnb >> 10);
        const int head  = __builtin_amdgcn_readfirstlane((gnb & 1023) >> 6);
        const int d     = (gnb & 63) + tl;
        const float bv  = bias[gnb + tl];
        const size_t bh = (size_t)(b * NHEAD + head);
        if (which == 0) {
            #pragma unroll
            for (int r = 0; r < 8; ++r)
                q16[(bh * SEQ + tb + r) * HDIM + d] = (_Float16)((acc[nt][r] + bv) * 0.125f);
        } else if (which == 1) {
            #pragma unroll
            for (int r = 0; r < 8; ++r)
                k16[(bh * SEQ + tb + r) * HDIM + d] = (_Float16)(acc[nt][r] + bv);
        } else {
            v8h pk;   // 8 consecutive t for fixed d -> single 16B store
            #pragma unroll
            for (int r = 0; r < 8; ++r) pk[r] = (_Float16)(acc[nt][r] + bv);
            *(v8h*)&vT16[(bh * HDIM + d) * SEQ + tb] = pk;
        }
    }
}

// ---------------------------------------------------------------------------
// Kernel 2: flash attention, transposed formulation.
// Per wave: 16 queries. S^T = K_blk @ q^T  (4 wmma / 32-key block),
// in-lane online softmax, O^T += V^T @ P^T (4 wmma / block; P^T fragment
// built in-lane from the S^T C-fragment: B-elem e = exp(sT[e>>3][e&7])).
// ---------------------------------------------------------------------------
__global__ __launch_bounds__(256) void attn_kernel(
    const _Float16* __restrict__ q16,
    const _Float16* __restrict__ k16,
    const _Float16* __restrict__ vT16,
    _Float16* __restrict__ y16)        // [B,T,C]
{
    const int tid  = threadIdx.x;
    const int lane = tid & 31;
    const int wave = tid >> 5;
    const int bh   = blockIdx.x;                    // 0..63
    const int t0   = blockIdx.y * 128 + wave * 16;  // query tile base
    const int b    = bh >> 4, head = bh & 15;
    const int tl   = lane & 15, hh = lane >> 4;
    const int t    = t0 + tl;

    // q^T B-fragments (D=64 -> two K=32 steps), loaded once
    const _Float16* qrow = q16 + ((size_t)bh * SEQ + t) * HDIM;
    v16h qB[2];
    #pragma unroll
    for (int ks = 0; ks < 2; ++ks)
        qB[ks] = frag_from(qrow + ks * 32 + 8 * hh);

    v8f oT[4];
    #pragma unroll
    for (int dt = 0; dt < 4; ++dt)
        #pragma unroll
        for (int r = 0; r < 8; ++r) oT[dt][r] = 0.0f;

    float mrun = -1e30f, lrun = 0.0f;
    const int nblk = ((t0 + 15) >> 5) + 1;

    for (int j = 0; j < nblk; ++j) {
        const int s0 = j * 32;

        if (s0 + 32 < SEQ) {   // prefetch next key block (global_prefetch_b8)
            __builtin_prefetch(k16 + ((size_t)bh * SEQ + (s0 + 32 + tl)) * HDIM, 0, 0);
        }

        // preload all K fragments (one load clause), then 4 wmma
        v16h kA[2][2];
        #pragma unroll
        for (int j2 = 0; j2 < 2; ++j2) {
            const _Float16* krow = k16 + ((size_t)bh * SEQ + (s0 + 16 * j2 + tl)) * HDIM;
            #pragma unroll
            for (int ks = 0; ks < 2; ++ks)
                kA[j2][ks] = frag_from(krow + ks * 32 + 8 * hh);
        }
        v8f sT[2];
        #pragma unroll
        for (int j2 = 0; j2 < 2; ++j2) {
            v8f z;
            #pragma unroll
            for (int r = 0; r < 8; ++r) z[r] = 0.0f;
            #pragma unroll
            for (int ks = 0; ks < 2; ++ks)
                z = __builtin_amdgcn_wmma_f32_16x16x32_f16(false, kA[j2][ks], false, qB[ks], (short)0, z, false, false);
            sT[j2] = z;
        }

        // causal mask + per-column (in-lane) max
        float mx = mrun;
        #pragma unroll
        for (int j2 = 0; j2 < 2; ++j2)
            #pragma unroll
            for (int r = 0; r < 8; ++r) {
                int s = s0 + 16 * j2 + r + 8 * hh;
                float v = (s <= t) ? sT[j2][r] : -1e30f;
                sT[j2][r] = v;
                mx = fmaxf(mx, v);
            }
        mx = fmaxf(mx, __shfl_xor(mx, 16, 32));   // merge lane-halves of column

        float alpha = __expf(mrun - mx);
        float psum = 0.0f;
        v16h pB;                                   // P^T B-fragment, built in-lane
        #pragma unroll
        for (int j2 = 0; j2 < 2; ++j2)
            #pragma unroll
            for (int r = 0; r < 8; ++r) {
                float p = __expf(sT[j2][r] - mx);
                psum += p;
                pB[j2 * 8 + r] = (_Float16)p;
            }
        psum += __shfl_xor(psum, 16, 32);
        lrun = lrun * alpha + psum;
        mrun = mx;

        // preload all V fragments, then O^T += V^T @ P^T  (4 d-tiles of 16)
        v16h vA[4];
        #pragma unroll
        for (int dt = 0; dt < 4; ++dt)
            vA[dt] = frag_from(vT16 + ((size_t)bh * HDIM + (16 * dt + tl)) * SEQ + s0 + 8 * hh);
        #pragma unroll
        for (int dt = 0; dt < 4; ++dt) {
            #pragma unroll
            for (int r = 0; r < 8; ++r) oT[dt][r] *= alpha;
            oT[dt] = __builtin_amdgcn_wmma_f32_16x16x32_f16(false, vA[dt], false, pB, (short)0, oT[dt], false, false);
        }
    }

    const float inv = 1.0f / lrun;
    _Float16* yrow = y16 + ((size_t)b * SEQ + t) * CDIM + head * HDIM;
    #pragma unroll
    for (int dt = 0; dt < 4; ++dt) {
        v8h outv;
        #pragma unroll
        for (int r = 0; r < 8; ++r) outv[r] = (_Float16)(oT[dt][r] * inv);
        *(v8h*)(yrow + dt * 16 + 8 * hh) = outv;
    }
}

// ---------------------------------------------------------------------------
// Kernel 3: out = y @ W_proj + b_proj   (M=8192, N=1024, K=1024), f32 out.
// Same 64x128 tile, double-buffered. A tile is already f16 -> staged with
// GLOBAL_LOAD_ASYNC_TO_LDS_B128 (ASYNCcnt path), no VGPR round trip.
// ---------------------------------------------------------------------------
__global__ __launch_bounds__(256) void proj_gemm_kernel(
    const _Float16* __restrict__ Y,   // [8192,1024] f16
    const float* __restrict__ W,      // [1024,1024]
    const float* __restrict__ bias,   // [1024]
    float* __restrict__ out)          // [8192,1024]
{
    __shared__ _Float16 As[2][64 * 32];
    __shared__ _Float16 Bs[2][128 * 32];

    const int tid  = threadIdx.x;
    const int lane = tid & 31;
    const int wave = tid >> 5;
    const int wm   = wave & 3;
    const int wn   = wave >> 2;
    const int m0   = blockIdx.x * 64;
    const int n0   = blockIdx.y * 128;
    const int tl   = lane & 15;
    const int hh   = lane >> 4;
    const int K    = 1024, ldb = 1024;

    v8f acc[4];
    #pragma unroll
    for (int nt = 0; nt < 4; ++nt)
        #pragma unroll
        for (int r = 0; r < 8; ++r) acc[nt][r] = 0.0f;

    const int am  = tid >> 2, akc = (tid & 3) * 8;
    const int bn  = tid & 127, bkc0 = (tid >> 7) * 16;
    const _Float16* aSrc = Y + (size_t)(m0 + am) * K + akc;
    const float*    bSrc = W + n0 + bn;

    auto stage = [&](int k0, int p) {
        // A tile: async global->LDS, 16B per lane, tracked by ASYNCcnt
        unsigned           ldsA = (unsigned)(uintptr_t)&As[p][am * 32 + akc];
        unsigned long long ga   = (unsigned long long)(uintptr_t)(aSrc + k0);
        asm volatile("global_load_async_to_lds_b128 %0, %1, off"
                     :: "v"(ldsA), "v"(ga) : "memory");
        #pragma unroll
        for (int c = 0; c < 2; ++c) {   // B tile: f32 -> f16 conversion staging
            int kc = bkc0 + c * 8;
            const float* s = bSrc + (size_t)(k0 + kc) * ldb;
            v8h pk;
            #pragma unroll
            for (int i = 0; i < 8; ++i) pk[i] = (_Float16)s[(size_t)i * ldb];
            *(v8h*)&Bs[p][bn * 32 + kc] = pk;
        }
    };

    stage(0, 0);
    for (int k0 = 0; k0 < K; k0 += 32) {
        const int p = (k0 >> 5) & 1;
        asm volatile("s_wait_asynccnt 0x0" ::: "memory");   // own async stores done
        __syncthreads();
        if (k0 + 32 < K) stage(k0 + 32, p ^ 1);

        v16h a = frag_from(&As[p][(wm * 16 + tl) * 32 + 8 * hh]);
        v16h bfr[4];
        #pragma unroll
        for (int nt = 0; nt < 4; ++nt)
            bfr[nt] = frag_from(&Bs[p][(wn * 64 + nt * 16 + tl) * 32 + 8 * hh]);
        #pragma unroll
        for (int nt = 0; nt < 4; ++nt)
            acc[nt] = __builtin_amdgcn_wmma_f32_16x16x32_f16(false, a, false, bfr[nt], (short)0, acc[nt], false, false);
    }

    #pragma unroll
    for (int nt = 0; nt < 4; ++nt) {
        const int gn = n0 + wn * 64 + nt * 16 + tl;
        const float bv = bias[gn];
        #pragma unroll
        for (int r = 0; r < 8; ++r) {
            const int gm = m0 + wm * 16 + r + 8 * hh;
            out[(size_t)gm * 1024 + gn] = acc[nt][r] + bv;
        }
    }
}

// ---------------------------------------------------------------------------
extern "C" void kernel_launch(void* const* d_in, const int* in_sizes, int n_in,
                              void* d_out, int out_size, void* d_ws, size_t ws_size,
                              hipStream_t stream) {
    (void)in_sizes; (void)n_in; (void)out_size; (void)ws_size;

    const float* x      = (const float*)d_in[0];
    const float* W_attn = (const float*)d_in[1];
    const float* b_attn = (const float*)d_in[2];
    const float* W_proj = (const float*)d_in[3];
    const float* b_proj = (const float*)d_in[4];
    float* out = (float*)d_out;

    const size_t qkvElems = (size_t)BATCH * NHEAD * SEQ * HDIM;   // 8,388,608
    _Float16* q16  = (_Float16*)d_ws;
    _Float16* k16  = q16 + qkvElems;
    _Float16* vT16 = k16 + qkvElems;
    _Float16* y16  = vT16 + qkvElems;                             // 16 MB each

    qkv_gemm_kernel<<<dim3(128, 24), 256, 0, stream>>>(x, W_attn, b_attn, q16, k16, vT16);
    attn_kernel<<<dim3(64, 16), 256, 0, stream>>>(q16, k16, vT16, y16);
    proj_gemm_kernel<<<dim3(128, 8), 256, 0, stream>>>(y16, W_proj, b_proj, out);
}
